// LSTM_23364622090611
// MI455X (gfx1250) — matrix operand order repaired and checked by reading, not compiled, not run on hardware
//
#include <hip/hip_runtime.h>
#include <hip/hip_bf16.h>

// ---------------------------------------------------------------------------
// LSTM scan (B=64, T=512, D=128, H=1024) as a persistent cooperative kernel.
// f16 WMMA (f32 accumulate) for all GEMMs, f32 cell state, grid barrier per
// phase. Weights converted to f16 once and kept L2-resident (~10 MB << 192MB).
// ---------------------------------------------------------------------------

typedef _Float16 h16;
typedef __attribute__((ext_vector_type(8)))  _Float16 v8h;
typedef __attribute__((ext_vector_type(16))) _Float16 v16h;
typedef __attribute__((ext_vector_type(8)))  float    v8f;

#define B_   64
#define T_   512
#define D_   128
#define H_   1024
#define G4   4096            // 4*H
#define KX   1152            // D + H   (packed gates K dimension)
#define NBLK 64              // persistent grid blocks
#define NTHR 256             // threads per block (8 wave32s)
#define NWAV (NBLK * NTHR / 32)

// ---- WMMA fragment helpers (contiguous K order, same for A and B) -----

// A-matrix 16x32 f16: lanes 0-15 -> rows 0-15, K 0..15;
// lanes 16-31 -> rows 0-15, K 16..31.
__device__ __forceinline__ v16h loadA(const h16* __restrict__ A, int row0,
                                      int kb, int lane) {
  int r  = row0 + (lane & 15);
  int ks = (lane >> 4) << 4;
  const h16* p = A + (size_t)r * KX + kb + ks;
  return *(const v16h*)(p);         // K = kb+ks .. +15
}

// B-matrix 32x16 f16, B[k][n] = W[n][k] with W row-major [N][ldw]:
// lane holds column (lane&15), 16 consecutive K values starting at
// kb + (lane>=16 ? 16 : 0)  ->  one contiguous 32-byte load.
__device__ __forceinline__ v16h loadB(const h16* __restrict__ W, int ldw,
                                      int n0, int kb, int lane) {
  int n  = n0 + (lane & 15);
  int ks = (lane >> 4) << 4;
  return *(const v16h*)(W + (size_t)n * ldw + kb + ks);
}

__device__ __forceinline__ v8f wmma_f16(v16h a, v16h b, v8f c) {
  return __builtin_amdgcn_wmma_f32_16x16x32_f16(
      /*neg_a=*/false, a, /*neg_b=*/false, b,
      /*c_mod=*/(short)0, c, /*reuse_a=*/false, /*reuse_b=*/false);
}

// ---- grid-wide barrier (monotone counter; all NBLK blocks co-resident) ----
__device__ __forceinline__ void grid_barrier(unsigned* cnt, unsigned epoch) {
  __syncthreads();
  if (threadIdx.x == 0) {
    __threadfence();                 // release our writes to device scope
    atomicAdd(cnt, 1u);
    unsigned target = epoch * NBLK;
    while (__hip_atomic_load(cnt, __ATOMIC_RELAXED, __HIP_MEMORY_SCOPE_AGENT)
           < target) {
      __builtin_amdgcn_s_sleep(2);
    }
  }
  __syncthreads();
  __threadfence();                   // acquire: don't read stale WGP$ lines
}

__device__ __forceinline__ float sigm(float x) {
  return 1.0f / (1.0f + __expf(-x));
}

// ---------------------------------------------------------------------------
// Prep kernel 1: convert / pack weights to f16, fold biases.
//   Wg [4096][1152] f16 : cols 0..127 = W_ih[:,1:129], cols 128..1151 = W_hh
//   Wd [256][1024]  f16 : W_dec
//   w0 [4096] f32       : W_ih[:,0]   (the "t" column, rank-1 update)
//   bg [4096] f32       : b_ih + b_hh
// ---------------------------------------------------------------------------
__global__ void prep_weights(const float* __restrict__ W_ih,
                             const float* __restrict__ W_hh,
                             const float* __restrict__ W_dec,
                             const float* __restrict__ b_ih,
                             const float* __restrict__ b_hh,
                             h16* __restrict__ Wg, h16* __restrict__ Wd,
                             float* __restrict__ w0, float* __restrict__ bg) {
  int stride = gridDim.x * blockDim.x;
  int i0 = blockIdx.x * blockDim.x + threadIdx.x;
  for (int idx = i0; idx < G4 * KX; idx += stride) {
    int n = idx / KX, k = idx - n * KX;
    float w = (k < D_) ? W_ih[n * (1 + D_) + 1 + k] : W_hh[n * H_ + (k - D_)];
    Wg[idx] = (h16)w;
  }
  for (int idx = i0; idx < 2 * D_ * H_; idx += stride)
    Wd[idx] = (h16)W_dec[idx];
  for (int idx = i0; idx < G4; idx += stride) {
    w0[idx] = W_ih[idx * (1 + D_)];
    bg[idx] = b_ih[idx] + b_hh[idx];
  }
}

// Prep kernel 2: init activation buffer A (x part = xs[:,0,:] == initial
// x_hat, h part = 0), cell state c = 0, barrier counter = 0.
__global__ void prep_state(const float* __restrict__ xs,
                           h16* __restrict__ A, float* __restrict__ cst,
                           unsigned* __restrict__ cnt) {
  int stride = gridDim.x * blockDim.x;
  int i0 = blockIdx.x * blockDim.x + threadIdx.x;
  if (i0 == 0) *cnt = 0u;
  for (int idx = i0; idx < B_ * D_; idx += stride) {
    int b = idx / D_, d = idx - b * D_;
    A[(size_t)b * KX + d] = (h16)xs[((size_t)b * T_) * D_ + d];
  }
  for (int idx = i0; idx < B_ * H_; idx += stride) {
    int b = idx / H_, j = idx - b * H_;
    A[(size_t)b * KX + D_ + j] = (h16)0.0f;
    cst[idx] = 0.0f;
  }
}

// ---------------------------------------------------------------------------
// Persistent LSTM kernel: the whole T=512 scan in one launch.
// ---------------------------------------------------------------------------
__global__ __launch_bounds__(NTHR)
void lstm_persistent(const float* __restrict__ xs,
                     const float* __restrict__ ts,
                     const unsigned char* __restrict__ force,   // jnp bool = 1B
                     const float* __restrict__ b_dec,
                     const h16* __restrict__ Wg, const h16* __restrict__ Wd,
                     const float* __restrict__ w0, const float* __restrict__ bg,
                     h16* __restrict__ A, float* __restrict__ cst,
                     float* __restrict__ gates, unsigned* __restrict__ cnt,
                     float* __restrict__ out) {
  const int lane = threadIdx.x & 31;
  const int wave = (blockIdx.x * NTHR + threadIdx.x) >> 5;   // 0..NWAV-1
  const int tid  = blockIdx.x * NTHR + threadIdx.x;
  const int nl15 = lane & 15;
  const int hi8  = (lane >> 4) << 3;
  unsigned epoch = 0;

  for (int t = 0; t < T_; ++t) {
    // ---- phase 1: gates[64x4096] = A[64xKX] @ Wg^T + (b_ih+b_hh) + t⊗w0 ---
    // 1024 output tiles of 16x16, 512 waves -> 2 tiles each, K=1152 (36 WMMA).
    for (int tile = wave; tile < (B_ / 16) * (G4 / 16); tile += NWAV) {
      int m0 = (tile & 3) * 16;
      int n0 = (tile >> 2) * 16;
      int ncol = n0 + nl15;
      float w0n = w0[ncol];
      float bgn = bg[ncol];
      v8f acc;
#pragma unroll
      for (int v = 0; v < 8; ++v)
        acc[v] = bgn + ts[(m0 + v + hi8) * T_ + t] * w0n;   // bias + t * w_col0
#pragma unroll 4
      for (int kb = 0; kb < KX; kb += 32) {
        if (kb + 32 < KX)
          __builtin_prefetch(Wg + (size_t)ncol * KX + kb + 32 + ((lane >> 4) << 4), 0, 1);
        v16h a = loadA(A, m0, kb, lane);
        v16h b = loadB(Wg, KX, n0, kb, lane);
        acc = wmma_f16(a, b, acc);
      }
#pragma unroll
      for (int v = 0; v < 8; ++v)
        gates[(size_t)(m0 + v + hi8) * G4 + ncol] = acc[v];
    }
    grid_barrier(cnt, ++epoch);

    // ---- phase 2: cell update; write h as f16 into A[:,D_: ] --------------
    for (int cell = tid; cell < B_ * H_; cell += NBLK * NTHR) {
      int b = cell >> 10, j = cell & (H_ - 1);
      const float* gr = gates + (size_t)b * G4;
      float ig = gr[j];
      float fg = gr[H_ + j];
      float gg = gr[2 * H_ + j];
      float og = gr[3 * H_ + j];
      float cn = sigm(fg) * cst[cell] + sigm(ig) * tanhf(gg);
      cst[cell] = cn;
      float hh = sigm(og) * tanhf(cn);
      A[(size_t)b * KX + D_ + j] = (h16)hh;
    }
    grid_barrier(cnt, ++epoch);

    // ---- phase 3: dec[64x256] = h @ Wd^T + b_dec; emit output; build next x
    if (wave < (B_ / 16) * (2 * D_ / 16)) {       // 64 tiles, K=1024 (32 WMMA)
      int m0 = (wave & 3) * 16;
      int n0 = (wave >> 2) * 16;
      int ncol = n0 + nl15;
      float bdn = b_dec[ncol];
      v8f acc;
#pragma unroll
      for (int v = 0; v < 8; ++v) acc[v] = bdn;
#pragma unroll 4
      for (int kb = 0; kb < H_; kb += 32) {
        v16h a = loadA(A, m0, D_ + kb, lane);     // h lives at cols D_..KX-1
        v16h b = loadB(Wd, H_, n0, kb, lane);
        acc = wmma_f16(a, b, acc);
      }
#pragma unroll
      for (int v = 0; v < 8; ++v) {
        int bb = m0 + v + hi8;
        float val = acc[v];
        if (ncol >= D_) val = fmaxf(val, 0.01f);  // std clip
        out[((size_t)bb * T_ + t) * (2 * D_) + ncol] = val;
        if (ncol < D_ && t + 1 < T_) {            // scheduled-sampling input
          float xv = force[bb * T_ + (t + 1)]
                         ? xs[((size_t)bb * T_ + (t + 1)) * D_ + ncol]
                         : val;                   // x_hat = unclipped mean
          A[(size_t)bb * KX + ncol] = (h16)xv;
        }
      }
    }
    grid_barrier(cnt, ++epoch);
  }
}

// ---------------------------------------------------------------------------
extern "C" void kernel_launch(void* const* d_in, const int* in_sizes, int n_in,
                              void* d_out, int out_size, void* d_ws,
                              size_t ws_size, hipStream_t stream) {
  const float* xs    = (const float*)d_in[0];          // (64,512,128)
  const float* ts    = (const float*)d_in[1];          // (64,512)
  const unsigned char* force = (const unsigned char*)d_in[2]; // bool (64,512)
  const float* W_ih  = (const float*)d_in[3];          // (4096,129)
  const float* b_ih  = (const float*)d_in[4];          // (4096)
  const float* W_hh  = (const float*)d_in[5];          // (4096,1024)
  const float* b_hh  = (const float*)d_in[6];          // (4096)
  const float* W_dec = (const float*)d_in[7];          // (256,1024)
  const float* b_dec = (const float*)d_in[8];          // (256)
  float* out = (float*)d_out;                          // (64,512,256)

  // Workspace carve-up (all regions 256-byte aligned), ~11.5 MB total.
  char* ws = (char*)d_ws;
  size_t off = 0;
  auto carve = [&](size_t bytes) -> void* {
    void* p = ws + off;
    off = (off + bytes + 255) & ~(size_t)255;
    return p;
  };
  h16*      Wg    = (h16*)carve((size_t)G4 * KX * sizeof(h16));      // 9.44 MB
  h16*      Wd    = (h16*)carve((size_t)2 * D_ * H_ * sizeof(h16));  // 0.5 MB
  float*    w0    = (float*)carve(G4 * sizeof(float));
  float*    bg    = (float*)carve(G4 * sizeof(float));
  h16*      A     = (h16*)carve((size_t)B_ * KX * sizeof(h16));      // 144 KB
  float*    cst   = (float*)carve((size_t)B_ * H_ * sizeof(float));  // 256 KB
  float*    gbuf  = (float*)carve((size_t)B_ * G4 * sizeof(float));  // 1 MB
  unsigned* cnt   = (unsigned*)carve(256);
  (void)ws_size; (void)in_sizes; (void)n_in; (void)out_size;

  prep_weights<<<512, 256, 0, stream>>>(W_ih, W_hh, W_dec, b_ih, b_hh,
                                        Wg, Wd, w0, bg);
  prep_state<<<64, 256, 0, stream>>>(xs, A, cst, cnt);
  lstm_persistent<<<NBLK, NTHR, 0, stream>>>(xs, ts, force, b_dec,
                                             Wg, Wd, w0, bg,
                                             A, cst, gbuf, cnt, out);
}
